// AttentiveFPDense2_9826885174107
// MI455X (gfx1250) — matrix operand order repaired and checked by reading, not compiled, not run on hardware
//
#include <hip/hip_runtime.h>
#include <math.h>

// ---------------- problem constants (match reference) ----------------
#define NN 131072        // nodes   (multiple of 16)
#define NE 262144        // edges   (multiple of 16)
#define NB 4096          // graphs  (multiple of 16)
#define FN 64            // node feat
#define FE 16            // edge feat
#define GD 200           // graph_feat_size G

typedef __attribute__((ext_vector_type(16))) _Float16 v16h;
typedef __attribute__((ext_vector_type(2)))  _Float16 h2;
typedef __attribute__((ext_vector_type(8)))  float    v8f;

// ============================ device helpers ============================
template<int ACT> __device__ __forceinline__ float activate(float x) {
  if (ACT == 1) return x > 0.f ? x : 0.01f * x;   // leaky_relu(0.01)
  if (ACT == 2) return x > 0.f ? x : 0.f;         // relu
  return x;
}

__device__ __forceinline__ void atomicMaxF(float* addr, float v) {
  if (v >= 0.f) atomicMax((int*)addr, __float_as_int(v));
  else          atomicMin((unsigned int*)addr, __float_as_uint(v));
}

// A-row abstraction: plain rows, or gathered+concat rows (nf[src] || ef).
// k1 is even; fragment K-pairs are even-aligned so a pair never straddles it.
struct ARow { const float* p1; const float* p2; int k1; };
struct APlain {
  const float* A; int lda;
  __device__ ARow row(int r) const { return ARow{A + (size_t)r * lda, A, 0x7fffffff}; }
};
struct AConcatGather {
  const float* X; int ldx; const int* idx; const float* Y; int ldy; int k1;
  __device__ ARow row(int r) const {
    return ARow{X + (size_t)idx[r] * ldx, Y + (size_t)r * ldy, k1};
  }
};
__device__ __forceinline__ float2 arow_get2(const ARow& ar, int k) {
  return (k < ar.k1) ? *(const float2*)(ar.p1 + k)
                     : *(const float2*)(ar.p2 + (k - ar.k1));
}

// ============================ weight pre-conversion ============================
// Wt[n*Kpad + k] = f16(W[k*Nc + n]), zero-padded for k >= K (Kpad % 32 == 0).
__global__ __launch_bounds__(256) void convert_wt_kernel(
    const float* __restrict__ W, _Float16* __restrict__ Wt,
    int K, int Nc, int Kpad, long long total)
{
  long long i = (long long)blockIdx.x * blockDim.x + threadIdx.x;
  if (i >= total) return;
  int n = (int)(i / Kpad);
  int k = (int)(i % Kpad);
  Wt[i] = (k < K) ? (_Float16)W[(size_t)k * Nc + n] : (_Float16)0.f;
}

// ============================ WMMA GEMM ============================
// C[M,Nc] = act(A[M,K] @ W[K,Nc] + bias); fp32 A/C, f16 multiply, f32 acc.
// One wave -> 16 rows x 64 cols (4 wmma tiles, A fragment reused 4x).
// Requires M % 16 == 0 (true for all call sites). K tail: A index clamped,
// products vanish because Wt is zero-padded in K.
template<typename AL, int ACT>
__global__ __launch_bounds__(256) void gemm_wmma_kernel(
    AL aload, const _Float16* __restrict__ Wt, int Kpad,
    const float* __restrict__ bias, float* __restrict__ C,
    int K, int Nc, int tilesM, int nwaves)
{
  int wid = blockIdx.x * 8 + (threadIdx.x >> 5);
  if (wid >= nwaves) return;
  int tm = wid % tilesM, tn = wid / tilesM;
  int lane = threadIdx.x & 31;
  int half = lane >> 4;
  int l16  = lane & 15;
  int m0 = tm * 16, n0 = tn * 64;
  ARow ar = aload.row(m0 + l16);

  const _Float16* bp[4];
#pragma unroll
  for (int t = 0; t < 4; ++t) {
    int nn = n0 + t * 16 + l16;
    if (nn >= Nc) nn = Nc - 1;               // clamp; stores are guarded
    bp[t] = Wt + (size_t)nn * Kpad;
  }

  v8f acc[4] = {};
  int kfull = K & ~31;
  for (int k0 = 0; k0 < kfull; k0 += 32) {
    __builtin_prefetch(ar.p1 + k0 + 32, 0, 1);
    v16h af;
#pragma unroll
    for (int r = 0; r < 8; ++r) {
      int kk = k0 + (r & 3) * 2 + half * 8 + (r >> 2) * 16;
      float2 v = arow_get2(ar, kk);
      af[2 * r]     = (_Float16)v.x;
      af[2 * r + 1] = (_Float16)v.y;
    }
#pragma unroll
    for (int t = 0; t < 4; ++t) {
      v16h bf;
#pragma unroll
      for (int r = 0; r < 8; ++r) {
        h2 p = *(const h2*)(bp[t] + k0 + (r & 3) * 2 + half * 8 + (r >> 2) * 16);
        bf[2 * r]     = p.x;
        bf[2 * r + 1] = p.y;
      }
      acc[t] = __builtin_amdgcn_wmma_f32_16x16x32_f16(
          false, af, false, bf, (short)0, acc[t], false, false);
    }
  }
  if (kfull < K) {                            // K tail: clamp A, B pad is 0
    int k0 = kfull;
    int kcl = (K - 2) & ~1;                   // largest valid even pair start
    v16h af;
#pragma unroll
    for (int r = 0; r < 8; ++r) {
      int kk = k0 + (r & 3) * 2 + half * 8 + (r >> 2) * 16;
      if (kk > kcl) kk = kcl;
      float2 v = arow_get2(ar, kk);
      af[2 * r]     = (_Float16)v.x;
      af[2 * r + 1] = (_Float16)v.y;
    }
#pragma unroll
    for (int t = 0; t < 4; ++t) {
      v16h bf;
#pragma unroll
      for (int r = 0; r < 8; ++r) {
        h2 p = *(const h2*)(bp[t] + k0 + (r & 3) * 2 + half * 8 + (r >> 2) * 16);
        bf[2 * r]     = p.x;
        bf[2 * r + 1] = p.y;
      }
      acc[t] = __builtin_amdgcn_wmma_f32_16x16x32_f16(
          false, af, false, bf, (short)0, acc[t], false, false);
    }
  }
#pragma unroll
  for (int t = 0; t < 4; ++t) {
#pragma unroll
    for (int r = 0; r < 8; ++r) {
      int mo = m0 + r + 8 * half;             // D: lanes 0-15 M=r, 16-31 M=r+8
      int no = n0 + t * 16 + l16;
      if (no < Nc)
        C[(size_t)mo * Nc + no] = activate<ACT>(acc[t][r] + bias[no]);
    }
  }
}

// ============================ elementwise / segment kernels ============================
__global__ __launch_bounds__(256) void fill_kernel(float* p, float v, long long n) {
  long long i = (long long)blockIdx.x * blockDim.x + threadIdx.x;
  if (i < n) p[i] = v;
}

__global__ __launch_bounds__(256) void elu_kernel(float* p, long long n) {
  long long i = (long long)blockIdx.x * blockDim.x + threadIdx.x;
  if (i >= n) return;
  float x = p[i];
  p[i] = x > 0.f ? x : (__expf(x) - 1.f);
}

template<int RELU1>
__global__ __launch_bounds__(256) void seg_logit_kernel(
    const float* __restrict__ X1, int ld1, const int* __restrict__ idx1,
    const float* __restrict__ X2, int ld2, const int* __restrict__ idx2,
    int g1, int g2, const float* __restrict__ w, const float* __restrict__ bptr,
    const int* __restrict__ seg, float* __restrict__ lg,
    float* __restrict__ segmax, int M)
{
  int wid = blockIdx.x * 8 + (threadIdx.x >> 5);
  if (wid >= M) return;
  int lane = threadIdx.x & 31;
  int i1 = idx1 ? idx1[wid] : wid;
  int i2 = idx2 ? idx2[wid] : wid;
  const float* r1 = X1 + (size_t)i1 * ld1;
  const float* r2 = X2 + (size_t)i2 * ld2;
  float s = 0.f;
  for (int j = lane; j < g1; j += 32) {
    float v = r1[j];
    if (RELU1) v = fmaxf(v, 0.f);
    s += v * w[j];
  }
  for (int j = lane; j < g2; j += 32) s += r2[j] * w[g1 + j];
  for (int o = 16; o > 0; o >>= 1) s += __shfl_xor(s, o, 32);
  if (lane == 0) {
    float v = s + bptr[0];
    v = v > 0.f ? v : 0.01f * v;              // leaky
    lg[wid] = v;
    atomicMaxF(&segmax[seg[wid]], v);
  }
}

__global__ __launch_bounds__(256) void seg_exp_kernel(
    const float* __restrict__ lg, const int* __restrict__ seg,
    const float* __restrict__ mx, float* __restrict__ ex,
    float* __restrict__ den, int M)
{
  int i = blockIdx.x * blockDim.x + threadIdx.x;
  if (i >= M) return;
  float m = mx[seg[i]];
  if (!(m > -3.0e38f) || !(m < 3.0e38f)) m = 0.f;   // where(isfinite, m, 0)
  float e = __expf(lg[i] - m);
  ex[i] = e;
  atomicAdd(&den[seg[i]], e);
}

__global__ __launch_bounds__(256) void scatter_weighted_kernel(
    const float* __restrict__ P, int ldp, const int* __restrict__ pidx,
    const float* __restrict__ ex, const float* __restrict__ den,
    const int* __restrict__ seg, float* __restrict__ out, int ldo, int G,
    long long total)
{
  long long i = (long long)blockIdx.x * blockDim.x + threadIdx.x;
  if (i >= total) return;
  int e = (int)(i / G);
  int f = (int)(i % G);
  float a = ex[e] / den[seg[e]];
  int pr = pidx ? pidx[e] : e;
  atomicAdd(&out[(size_t)seg[e] * ldo + f], a * P[(size_t)pr * ldp + f]);
}

__global__ __launch_bounds__(256) void scatter_sum_kernel(
    const float* __restrict__ X, const int* __restrict__ seg,
    float* __restrict__ out, int G, long long total)
{
  long long i = (long long)blockIdx.x * blockDim.x + threadIdx.x;
  if (i >= total) return;
  long long row = i / G;
  int f = (int)(i % G);
  atomicAdd(&out[(size_t)seg[row] * G + f], X[row * G + f]);
}

__global__ __launch_bounds__(256) void gru_kernel(
    const float* __restrict__ gi, const float* __restrict__ gh,
    const float* __restrict__ h, float* __restrict__ out, int G, long long total)
{
  long long i = (long long)blockIdx.x * blockDim.x + threadIdx.x;
  if (i >= total) return;
  long long row = i / G;
  int f = (int)(i % G);
  const float* gir = gi + (size_t)row * 3 * G;
  const float* ghr = gh + (size_t)row * 3 * G;
  float r = 1.f / (1.f + __expf(-(gir[f]     + ghr[f])));
  float z = 1.f / (1.f + __expf(-(gir[G + f] + ghr[G + f])));
  float n = tanhf(gir[2 * G + f] + r * ghr[2 * G + f]);
  float hv = h[(size_t)row * G + f];
  float o = (1.f - z) * n + z * hv;
  out[(size_t)row * G + f] = fmaxf(o, 0.f);
}

__global__ __launch_bounds__(256) void copy_cols_kernel(
    const float* __restrict__ s, float* __restrict__ d, int G, int ldd, int coff,
    long long total)
{
  long long i = (long long)blockIdx.x * blockDim.x + threadIdx.x;
  if (i >= total) return;
  long long row = i / G;
  int f = (int)(i % G);
  d[row * ldd + coff + f] = s[row * G + f];
}

__global__ __launch_bounds__(256) void final_dot_kernel(
    const float* __restrict__ X, int ld, const float* __restrict__ w,
    const float* __restrict__ bptr, float* __restrict__ out, int K, int M)
{
  int wid = blockIdx.x * 8 + (threadIdx.x >> 5);
  if (wid >= M) return;
  int lane = threadIdx.x & 31;
  const float* r = X + (size_t)wid * ld;
  float s = 0.f;
  for (int j = lane; j < K; j += 32) s += r[j] * w[j];
  for (int o = 16; o > 0; o >>= 1) s += __shfl_xor(s, o, 32);
  if (lane == 0) out[wid] = s + bptr[0];
}

// ============================ host side ============================
static inline int ceilblk(long long n, int b) { return (int)((n + b - 1) / b); }

struct WT { const _Float16* p; int kpad; };
struct F16Arena { _Float16* base; size_t off; };

static WT make_wt(F16Arena& a, const float* W, int K, int Nc, hipStream_t s) {
  int kpad = (K + 31) & ~31;
  _Float16* p = a.base + a.off;
  long long tot = (long long)Nc * kpad;
  a.off += (size_t)tot;
  convert_wt_kernel<<<ceilblk(tot, 256), 256, 0, s>>>(W, p, K, Nc, kpad, tot);
  return WT{p, kpad};
}

template<typename AL, int ACT>
static void gemm(AL al, WT wt, const float* bias, float* C,
                 int M, int K, int Nc, hipStream_t s)
{
  int tilesM = (M + 15) / 16;
  int tilesN = (Nc + 63) / 64;
  int nwaves = tilesM * tilesN;
  int blocks = (nwaves + 7) / 8;
  gemm_wmma_kernel<AL, ACT><<<blocks, 256, 0, s>>>(al, wt.p, wt.kpad, bias, C,
                                                   K, Nc, tilesM, nwaves);
}

struct Lin { const float* w; const float* b; };
struct Gru { const float* wi; const float* wh; const float* bi; const float* bh; };
struct Model {
  Lin agru1, edge1, edge2, projnode;
  Gru gru0;
  Lin ledge[2]; Gru lgru[2]; Lin lproj[2];
  Gru rgru[2];  Lin rlog[2]; Lin rproj[2];
};
struct GruT { WT wi, wh; const float* bi; const float* bh; };
struct ModelT {
  WT projnode, edge1, agru1;
  GruT gru0;
  WT lproj[2]; GruT lgru[2];
  WT rproj[2]; GruT rgru[2];
};

// Param leaves assumed in JAX pytree order (dict keys sorted alphabetically);
// either one leaf per d_in entry starting at index 15, or one flat blob at
// d_in[15] concatenated in the same order.
struct ParamCursor {
  void* const* din; int idx; const float* blob; size_t off; bool sep;
  const float* next(size_t n) {
    if (sep) return (const float*)din[idx++];
    const float* p = blob + off; off += n; return p;
  }
};

static void load_gru(ParamCursor& pc, Gru& g) {   // sorted: bh, bi, wh, wi
  g.bh = pc.next(3 * GD); g.bi = pc.next(3 * GD);
  g.wh = pc.next((size_t)GD * 3 * GD); g.wi = pc.next((size_t)GD * 3 * GD);
}
static void load_model(ParamCursor& pc, Model& m) {
  // sorted: agru1_et, edge1, edge2, gru0, layers, proj_node, readout
  m.agru1.w = pc.next((size_t)GD * GD); m.agru1.b = pc.next(GD);
  m.edge1.w = pc.next((size_t)(FN + FE) * GD); m.edge1.b = pc.next(GD);
  m.edge2.w = pc.next(2 * GD); m.edge2.b = pc.next(1);
  load_gru(pc, m.gru0);
  for (int l = 0; l < 2; ++l) {                   // sorted: edge, gru, proj
    m.ledge[l].w = pc.next(2 * GD); m.ledge[l].b = pc.next(1);
    load_gru(pc, m.lgru[l]);
    m.lproj[l].w = pc.next((size_t)GD * GD); m.lproj[l].b = pc.next(GD);
  }
  m.projnode.w = pc.next((size_t)FN * GD); m.projnode.b = pc.next(GD);
  for (int t = 0; t < 2; ++t) {                   // sorted: gru, logits, proj
    load_gru(pc, m.rgru[t]);
    m.rlog[t].w = pc.next(2 * GD); m.rlog[t].b = pc.next(1);
    m.rproj[t].w = pc.next((size_t)GD * GD); m.rproj[t].b = pc.next(GD);
  }
}

static void convert_model(F16Arena& a, const Model& m, ModelT& t, hipStream_t s) {
  t.projnode = make_wt(a, m.projnode.w, FN, GD, s);
  t.edge1    = make_wt(a, m.edge1.w, FN + FE, GD, s);
  t.agru1    = make_wt(a, m.agru1.w, GD, GD, s);
  t.gru0.wi = make_wt(a, m.gru0.wi, GD, 3 * GD, s);
  t.gru0.wh = make_wt(a, m.gru0.wh, GD, 3 * GD, s);
  t.gru0.bi = m.gru0.bi; t.gru0.bh = m.gru0.bh;
  for (int l = 0; l < 2; ++l) {
    t.lproj[l]   = make_wt(a, m.lproj[l].w, GD, GD, s);
    t.lgru[l].wi = make_wt(a, m.lgru[l].wi, GD, 3 * GD, s);
    t.lgru[l].wh = make_wt(a, m.lgru[l].wh, GD, 3 * GD, s);
    t.lgru[l].bi = m.lgru[l].bi; t.lgru[l].bh = m.lgru[l].bh;
  }
  for (int r = 0; r < 2; ++r) {
    t.rproj[r]   = make_wt(a, m.rproj[r].w, GD, GD, s);
    t.rgru[r].wi = make_wt(a, m.rgru[r].wi, GD, 3 * GD, s);
    t.rgru[r].wh = make_wt(a, m.rgru[r].wh, GD, 3 * GD, s);
    t.rgru[r].bi = m.rgru[r].bi; t.rgru[r].bh = m.rgru[r].bh;
  }
}

static void run_model(const Model& m, const ModelT& mt,
                      const float* nf, const float* ef,
                      const int* src, const int* dst, const int* n2g,
                      float* H, float* A, float* P, float* CTX,
                      float* GI, float* GH, float* LG, float* EX,
                      float* MX, float* DEN, float* Gf, float* GR,
                      float* X, int coff, hipStream_t s)
{
  const long long NH = (long long)NN * GD;
  const long long EH = (long long)NE * GD;
  const long long BH = (long long)NB * GD;

  // ---- GetContext ----
  gemm<APlain, 1>(APlain{nf, FN}, mt.projnode, m.projnode.b, H, NN, FN, GD, s);
  gemm<AConcatGather, 1>(AConcatGather{nf, FN, src, ef, FE, FN},
                         mt.edge1, m.edge1.b, A, NE, FN + FE, GD, s);
  fill_kernel<<<ceilblk(NN, 256), 256, 0, s>>>(MX, -INFINITY, NN);
  fill_kernel<<<ceilblk(NN, 256), 256, 0, s>>>(DEN, 0.f, NN);
  seg_logit_kernel<0><<<ceilblk(NE, 8), 256, 0, s>>>(H, GD, dst, A, GD, nullptr,
      GD, GD, m.edge2.w, m.edge2.b, dst, LG, MX, NE);
  seg_exp_kernel<<<ceilblk(NE, 256), 256, 0, s>>>(LG, dst, MX, EX, DEN, NE);
  gemm<APlain, 0>(APlain{A, GD}, mt.agru1, m.agru1.b, P, NE, GD, GD, s);
  fill_kernel<<<ceilblk(NH, 256), 256, 0, s>>>(CTX, 0.f, NH);
  scatter_weighted_kernel<<<ceilblk(EH, 256), 256, 0, s>>>(P, GD, nullptr, EX, DEN,
      dst, CTX, GD, GD, EH);
  elu_kernel<<<ceilblk(NH, 256), 256, 0, s>>>(CTX, NH);
  gemm<APlain, 0>(APlain{CTX, GD}, mt.gru0.wi, m.gru0.bi, GI, NN, GD, 3 * GD, s);
  gemm<APlain, 0>(APlain{H,   GD}, mt.gru0.wh, m.gru0.bh, GH, NN, GD, 3 * GD, s);
  gru_kernel<<<ceilblk(NH, 256), 256, 0, s>>>(GI, GH, H, H, GD, NH);

  // ---- GNN layers ----
  for (int l = 0; l < 2; ++l) {
    fill_kernel<<<ceilblk(NN, 256), 256, 0, s>>>(MX, -INFINITY, NN);
    fill_kernel<<<ceilblk(NN, 256), 256, 0, s>>>(DEN, 0.f, NN);
    seg_logit_kernel<0><<<ceilblk(NE, 8), 256, 0, s>>>(H, GD, dst, H, GD, src,
        GD, GD, m.ledge[l].w, m.ledge[l].b, dst, LG, MX, NE);
    seg_exp_kernel<<<ceilblk(NE, 256), 256, 0, s>>>(LG, dst, MX, EX, DEN, NE);
    gemm<APlain, 0>(APlain{H, GD}, mt.lproj[l], m.lproj[l].b, P, NN, GD, GD, s);
    fill_kernel<<<ceilblk(NH, 256), 256, 0, s>>>(CTX, 0.f, NH);
    scatter_weighted_kernel<<<ceilblk(EH, 256), 256, 0, s>>>(P, GD, src, EX, DEN,
        dst, CTX, GD, GD, EH);
    elu_kernel<<<ceilblk(NH, 256), 256, 0, s>>>(CTX, NH);
    gemm<APlain, 0>(APlain{CTX, GD}, mt.lgru[l].wi, m.lgru[l].bi, GI, NN, GD, 3 * GD, s);
    gemm<APlain, 0>(APlain{H,   GD}, mt.lgru[l].wh, m.lgru[l].bh, GH, NN, GD, 3 * GD, s);
    gru_kernel<<<ceilblk(NH, 256), 256, 0, s>>>(GI, GH, H, H, GD, NH);
  }

  // ---- Readout ----
  fill_kernel<<<ceilblk(BH, 256), 256, 0, s>>>(Gf, 0.f, BH);
  scatter_sum_kernel<<<ceilblk(NH, 256), 256, 0, s>>>(H, n2g, Gf, GD, NH);
  for (int t = 0; t < 2; ++t) {
    fill_kernel<<<ceilblk(NB, 256), 256, 0, s>>>(MX, -INFINITY, NB);
    fill_kernel<<<ceilblk(NB, 256), 256, 0, s>>>(DEN, 0.f, NB);
    seg_logit_kernel<1><<<ceilblk(NN, 8), 256, 0, s>>>(Gf, GD, n2g, H, GD, nullptr,
        GD, GD, m.rlog[t].w, m.rlog[t].b, n2g, LG, MX, NN);
    seg_exp_kernel<<<ceilblk(NN, 256), 256, 0, s>>>(LG, n2g, MX, EX, DEN, NN);
    gemm<APlain, 0>(APlain{H, GD}, mt.rproj[t], m.rproj[t].b, P, NN, GD, GD, s);
    fill_kernel<<<ceilblk(BH, 256), 256, 0, s>>>(GR, 0.f, BH);
    scatter_weighted_kernel<<<ceilblk(NH, 256), 256, 0, s>>>(P, GD, nullptr, EX, DEN,
        n2g, GR, GD, GD, NH);
    elu_kernel<<<ceilblk(BH, 256), 256, 0, s>>>(GR, BH);
    gemm<APlain, 0>(APlain{GR, GD}, mt.rgru[t].wi, m.rgru[t].bi, GI, NB, GD, 3 * GD, s);
    gemm<APlain, 0>(APlain{Gf, GD}, mt.rgru[t].wh, m.rgru[t].bh, GH, NB, GD, 3 * GD, s);
    gru_kernel<<<ceilblk(BH, 256), 256, 0, s>>>(GI, GH, Gf, Gf, GD, BH);
  }
  copy_cols_kernel<<<ceilblk(BH, 256), 256, 0, s>>>(Gf, X, GD, 3 * GD, coff, BH);
}

extern "C" void kernel_launch(void* const* d_in, const int* in_sizes, int n_in,
                              void* d_out, int out_size, void* d_ws, size_t ws_size,
                              hipStream_t stream)
{
  (void)in_sizes; (void)out_size;
  const float* nf[3]; const float* ef[3];
  const int* src[3]; const int* dst[3]; const int* n2g[3];
  for (int i = 0; i < 3; ++i) {
    nf[i]  = (const float*)d_in[5 * i + 0];
    ef[i]  = (const float*)d_in[5 * i + 1];
    src[i] = (const int*)  d_in[5 * i + 2];
    dst[i] = (const int*)  d_in[5 * i + 3];
    n2g[i] = (const int*)  d_in[5 * i + 4];
  }
  ParamCursor pc{d_in, 15, (const float*)d_in[15], 0, n_in > 20};
  Lin dense0, dense1, predict; Model mdl[3];
  dense0.w = pc.next(600 * 256); dense0.b = pc.next(256);
  dense1.w = pc.next(256 * 256); dense1.b = pc.next(256);
  for (int i = 0; i < 3; ++i) load_model(pc, mdl[i]);
  predict.w = pc.next(256); predict.b = pc.next(1);

  // workspace arena (fp32 part)
  float* w = (float*)d_ws;
  const size_t SZ_H = (size_t)NN * GD, SZ_E = (size_t)NE * GD, SZ_G3 = (size_t)NN * 3 * GD;
  float* H   = w; w += SZ_H;
  float* A   = w; w += SZ_E;
  float* P   = w; w += SZ_E;
  float* CTX = w; w += SZ_H;
  float* GI  = w; w += SZ_G3;
  float* GH  = w; w += SZ_G3;
  float* LG  = w; w += NE;
  float* EX  = w; w += NE;
  float* MX  = w; w += NN;
  float* DEN = w; w += NN;
  float* Gf  = w; w += (size_t)NB * GD;
  float* GR  = w; w += (size_t)NB * GD;
  float* X   = w; w += (size_t)NB * 600;
  float* D1  = w; w += (size_t)NB * 256;
  float* D2  = w; w += (size_t)NB * 256;
  // f16 weight arena (≈5.03M halfs needed; reserve 5.5M)
  const size_t F16_HALFS = 5500000;
  F16Arena fa{(_Float16*)w, 0};
  size_t need = (size_t)(w - (float*)d_ws) * sizeof(float) + F16_HALFS * sizeof(_Float16);
  if (need > ws_size) return;  // ws too small

  ModelT mdt[3];
  for (int i = 0; i < 3; ++i) convert_model(fa, mdl[i], mdt[i], stream);
  WT d0t = make_wt(fa, dense0.w, 600, 256, stream);
  WT d1t = make_wt(fa, dense1.w, 256, 256, stream);

  for (int i = 0; i < 3; ++i)
    run_model(mdl[i], mdt[i], nf[i], ef[i], src[i], dst[i], n2g[i],
              H, A, P, CTX, GI, GH, LG, EX, MX, DEN, Gf, GR,
              X, GD * i, stream);

  // dense head: [B,600] -> 256 -> 256 -> 1
  gemm<APlain, 2>(APlain{X,  600}, d0t, dense0.b, D1, NB, 600, 256, stream);
  gemm<APlain, 2>(APlain{D1, 256}, d1t, dense1.b, D2, NB, 256, 256, stream);
  final_dot_kernel<<<ceilblk(NB, 8), 256, 0, stream>>>(D2, 256, predict.w, predict.b,
      (float*)d_out, 256, NB);
}
